// NMT_54176717472286
// MI455X (gfx1250) — matrix operand order repaired and checked by reading, not compiled, not run on hardware
//
#include <hip/hip_runtime.h>
#include <math.h>

#define SS 64      // src len
#define BB 32      // batch
#define TT 64      // trg len
#define EE 512     // embed
#define HH 1024    // decoder hidden
#define HE 512     // per-direction encoder hidden
#define VT 32000   // trg vocab
#define START_TOK 2

typedef __attribute__((ext_vector_type(2)))  float  v2f;
typedef __attribute__((ext_vector_type(8)))  float  v8f;
typedef __attribute__((ext_vector_type(8)))  __bf16 v8bf;
typedef __attribute__((ext_vector_type(16))) __bf16 v16bf;

static __device__ __forceinline__ __bf16 f2bf(float f) {
  unsigned u = __builtin_bit_cast(unsigned, f);
  unsigned r = u + 0x7FFFu + ((u >> 16) & 1u);       // round-to-nearest-even
  unsigned short h = (unsigned short)(r >> 16);
  return __builtin_bit_cast(__bf16, h);
}
static __device__ __forceinline__ float sigm(float x) { return 1.0f / (1.0f + expf(-x)); }

// ---------------------------------------------------------------------------
// Generic fp32 WMMA GEMM: C[m][n] = act( Cin[m][n] + bias[n] + sum_k A[row(m)][k]*B[k][n] )
//   A rows optionally gathered via rowidx (embedding lookup fused into GEMM).
//   W stored [N, ldw] (wtrans=0, B[k][n]=W[n][wcol0+k]) or [K, ldw] (wtrans=1,
//   B[k][n]=W[wcol0+k][n]).  One wave per 16x16 C tile, K-loop of v_wmma_f32_16x16x4_f32.
// ---------------------------------------------------------------------------
__global__ void gemm_f32_wmma(
    const float* __restrict__ A, int lda, const int* __restrict__ rowidx,
    const float* __restrict__ W, int ldw, int wcol0, int wtrans,
    const float* __restrict__ bias,
    const float* __restrict__ Cin, int ldcin,
    float* __restrict__ C, int ldc, int K, int act)
{
  const int tn = blockIdx.x << 4;
  const int tm = blockIdx.y << 4;
  const int lane = threadIdx.x;
  const int lo = lane & 15;
  const int hi = lane >> 4;

  const int am   = tm + lo;
  const long arow = rowidx ? (long)rowidx[am] : (long)am;
  const float* Ar = A + arow * lda;

  v8f acc = {};
  if (wtrans) {
    for (int k0 = 0; k0 < K; k0 += 4) {
      v2f a, b;
      a.x = Ar[k0 + 2*hi];
      a.y = Ar[k0 + 2*hi + 1];
      b.x = W[(long)(wcol0 + k0 + 2*hi)     * ldw + tn + lo];
      b.y = W[(long)(wcol0 + k0 + 2*hi + 1) * ldw + tn + lo];
      acc = __builtin_amdgcn_wmma_f32_16x16x4_f32(false, a, false, b, (short)0, acc, false, false);
    }
  } else {
    const float* Wr = W + (long)(tn + lo) * ldw + wcol0;
    for (int k0 = 0; k0 < K; k0 += 4) {
      v2f a, b;
      a.x = Ar[k0 + 2*hi];
      a.y = Ar[k0 + 2*hi + 1];
      b.x = Wr[k0 + 2*hi];
      b.y = Wr[k0 + 2*hi + 1];
      acc = __builtin_amdgcn_wmma_f32_16x16x4_f32(false, a, false, b, (short)0, acc, false, false);
    }
  }

  const int col = tn + lo;
  const float bv = bias ? bias[col] : 0.0f;
#pragma unroll
  for (int r = 0; r < 8; ++r) {
    const int row = tm + r + 8*hi;
    float v = acc[r] + bv;
    if (Cin) v += Cin[(long)row * ldcin + col];
    if (act == 1) v = tanhf(v);
    C[(long)row * ldc + col] = v;
  }
}

// ---------------------------------------------------------------------------
// Output projection, pre-converted bf16 operands (preferred path):
// logits = cc16 @ Wout16^T + bout via v_wmma_f32_16x16x32_bf16, fp32 accum.
// Pure vector loads + WMMA in the loop; Wout stream prefetched.
// ---------------------------------------------------------------------------
__global__ void outproj_bf16pre_wmma(
    const __bf16* __restrict__ A, const __bf16* __restrict__ W,
    const float* __restrict__ bias, float* __restrict__ C, int ldc, int K)
{
  const int tn = blockIdx.x << 4;
  const int tm = blockIdx.y << 4;
  const int lane = threadIdx.x;
  const int lo = lane & 15;
  const int hi = lane >> 4;

  const __bf16* Ar = A + (long)(tm + lo) * K;
  const __bf16* Wr = W + (long)(tn + lo) * K;

  v8f acc = {};
  for (int k0 = 0; k0 < K; k0 += 32) {
    __builtin_prefetch(Wr + k0 + 256, 0, 1);         // global_prefetch_b8: next Wout chunk
    const v8bf alo = *(const v8bf*)(Ar + k0 + 8*hi);        // K = 8*hi + 0..7
    const v8bf ahi = *(const v8bf*)(Ar + k0 + 16 + 8*hi);   // K = 16 + 8*hi + 0..7
    const v16bf a = __builtin_shufflevector(alo, ahi,
        0,1,2,3,4,5,6,7,8,9,10,11,12,13,14,15);
    const v8bf blo = *(const v8bf*)(Wr + k0 + 8*hi);        // K = 8*hi + 0..7
    const v8bf bhi = *(const v8bf*)(Wr + k0 + 16 + 8*hi);   // K = 16 + 8*hi + 0..7
    const v16bf b = __builtin_shufflevector(blo, bhi,
        0,1,2,3,4,5,6,7,8,9,10,11,12,13,14,15);
    acc = __builtin_amdgcn_wmma_f32_16x16x32_bf16(false, a, false, b, (short)0, acc, false, false);
  }

  const int col = tn + lo;
  const float bv = bias[col];
#pragma unroll
  for (int r = 0; r < 8; ++r) {
    const int row = tm + r + 8*hi;
    C[(long)row * ldc + col] = acc[r] + bv;
  }
}

// Fallback: same GEMM with on-the-fly fp32->bf16 conversion (used only if the
// workspace cannot hold a bf16 copy of Wout).
__global__ void outproj_bf16_wmma(
    const float* __restrict__ A, const float* __restrict__ W,
    const float* __restrict__ bias, float* __restrict__ C, int ldc, int K)
{
  const int tn = blockIdx.x << 4;
  const int tm = blockIdx.y << 4;
  const int lane = threadIdx.x;
  const int lo = lane & 15;
  const int hi = lane >> 4;

  const float* Ar = A + (long)(tm + lo) * K;
  const float* Wr = W + (long)(tn + lo) * K;

  v8f acc = {};
  for (int k0 = 0; k0 < K; k0 += 32) {
    v16bf a, b;
#pragma unroll
    for (int j = 0; j < 8; ++j) {
      a[j]     = f2bf(Ar[k0 + 8*hi + j]);
      a[j + 8] = f2bf(Ar[k0 + 16 + 8*hi + j]);
    }
#pragma unroll
    for (int j = 0; j < 16; ++j)
      b[j] = f2bf(Wr[k0 + 8*hi + (j & 7) + ((j >> 3) << 4)]);
    acc = __builtin_amdgcn_wmma_f32_16x16x32_bf16(false, a, false, b, (short)0, acc, false, false);
  }

  const int col = tn + lo;
  const float bv = bias[col];
#pragma unroll
  for (int r = 0; r < 8; ++r) {
    const int row = tm + r + 8*hi;
    C[(long)row * ldc + col] = acc[r] + bv;
  }
}

// streaming fp32 -> bf16 conversion
__global__ void f32_to_bf16(const float* __restrict__ src, __bf16* __restrict__ dst, long n)
{
  long i = (long)blockIdx.x * blockDim.x + threadIdx.x;
  const long stride = (long)gridDim.x * blockDim.x;
  for (; i < n; i += stride) dst[i] = f2bf(src[i]);
}

// ---------------------------------------------------------------------------
// Elementwise LSTM cell (PyTorch gate order i,f,g,o), optional extra h sink
// (encoder output tensor laid out [B,S,H]).
// ---------------------------------------------------------------------------
__global__ void lstm_cell(const float* __restrict__ g, int ldg, int Hc,
                          float* __restrict__ h, float* __restrict__ c,
                          float* __restrict__ hout2, long hout2_bstride)
{
  const int idx = blockIdx.x * blockDim.x + threadIdx.x;
  if (idx >= BB * Hc) return;
  const int b = idx / Hc, j = idx % Hc;
  const float* gr = g + (long)b * ldg;
  const float gi = gr[j], gf = gr[Hc + j], gg = gr[2*Hc + j], go = gr[3*Hc + j];
  const float cn = sigm(gf) * c[idx] + sigm(gi) * tanhf(gg);
  const float hn = sigm(go) * tanhf(cn);
  c[idx] = cn;
  h[idx] = hn;
  if (hout2) hout2[(long)b * hout2_bstride + j] = hn;
}

// score[b,s] = h[b,:] . keys[b,s,:]   (one wave per (b,s), wave32 reduce)
__global__ void attn_score(const float* __restrict__ h, const float* __restrict__ keys,
                           float* __restrict__ score)
{
  const int m = blockIdx.x;              // b*S + s
  const int b = m / SS;
  const int lane = threadIdx.x;
  const float* hr = h + (long)b * HH;
  const float* kr = keys + (long)m * HH;
  float s = 0.0f;
  for (int j = lane; j < HH; j += 32) s += hr[j] * kr[j];
#pragma unroll
  for (int off = 16; off > 0; off >>= 1) s += __shfl_xor(s, off, 32);
  if (lane == 0) score[m] = s;
}

// softmax over S, then ctx[b,:] = sum_s a[b,s] * enc[b,s,:]
__global__ void softmax_ctx(const float* __restrict__ score, const float* __restrict__ enc,
                            float* __restrict__ ctx)
{
  __shared__ float a[SS];
  __shared__ float inv_s;
  const int b = blockIdx.x, tid = threadIdx.x;
  if (tid == 0) {
    float mx = -1e30f;
    for (int s = 0; s < SS; ++s) mx = fmaxf(mx, score[b*SS + s]);
    float sum = 0.0f;
    for (int s = 0; s < SS; ++s) { float e = expf(score[b*SS + s] - mx); a[s] = e; sum += e; }
    inv_s = 1.0f / sum;
  }
  __syncthreads();
  const float inv = inv_s;
  for (int hh = tid; hh < HH; hh += blockDim.x) {
    float acc = 0.0f;
    for (int s = 0; s < SS; ++s) acc += a[s] * enc[((long)b*SS + s)*HH + hh];
    ctx[(long)b*HH + hh] = acc * inv;
  }
}

// in-place log_softmax over n per row (block per row)
__global__ void log_softmax_rows(float* __restrict__ x, int n)
{
  __shared__ float red[256];
  float* row = x + (long)blockIdx.x * n;
  const int tid = threadIdx.x;
  float mx = -1e30f;
  for (int v = tid; v < n; v += 256) mx = fmaxf(mx, row[v]);
  red[tid] = mx; __syncthreads();
  for (int o = 128; o > 0; o >>= 1) { if (tid < o) red[tid] = fmaxf(red[tid], red[tid + o]); __syncthreads(); }
  mx = red[0]; __syncthreads();
  float sum = 0.0f;
  for (int v = tid; v < n; v += 256) sum += expf(row[v] - mx);
  red[tid] = sum; __syncthreads();
  for (int o = 128; o > 0; o >>= 1) { if (tid < o) red[tid] += red[tid + o]; __syncthreads(); }
  const float lse = mx + logf(red[0]);
  for (int v = tid; v < n; v += 256) row[v] -= lse;
}

// token index tables: fwd src, reversed src, decoder teacher-forced input
__global__ void build_idx(const int* __restrict__ src, const int* __restrict__ trg,
                          int* __restrict__ idxf, int* __restrict__ idxb, int* __restrict__ idxd)
{
  const int i = blockIdx.x * blockDim.x + threadIdx.x;
  if (i < SS * BB) {
    const int t = i / BB, b = i % BB;
    idxf[i] = src[i];
    idxb[i] = src[(SS - 1 - t) * BB + b];
    idxd[i] = (t == 0) ? START_TOK : trg[(t - 1) * BB + b];
  }
}

// out[b,0:H] = {hf[b,0:HE], hb[b,0:HE]}
__global__ void concat_hc(const float* __restrict__ hf, const float* __restrict__ hb,
                          float* __restrict__ out)
{
  const int i = blockIdx.x * blockDim.x + threadIdx.x;
  if (i < BB * HH) {
    const int b = i / HH, j = i % HH;
    out[i] = (j < HE) ? hf[b*HE + j] : hb[b*HE + (j - HE)];
  }
}

// ---------------------------------------------------------------------------
extern "C" void kernel_launch(void* const* d_in, const int* in_sizes, int n_in,
                              void* d_out, int out_size, void* d_ws, size_t ws_size,
                              hipStream_t stream) {
  (void)in_sizes; (void)n_in; (void)out_size;
  const int*   src_batch = (const int*)  d_in[0];
  const int*   trg_batch = (const int*)  d_in[1];
  const float* emb_src   = (const float*)d_in[2];
  const float* eWih_f    = (const float*)d_in[3];
  const float* eWhh_f    = (const float*)d_in[4];
  const float* eb_f      = (const float*)d_in[5];
  const float* eWih_b    = (const float*)d_in[6];
  const float* eWhh_b    = (const float*)d_in[7];
  const float* eb_b      = (const float*)d_in[8];
  const float* emb_trg   = (const float*)d_in[9];
  const float* dWih      = (const float*)d_in[10];  // [4H, E+H]
  const float* dWhh      = (const float*)d_in[11];  // [4H, H]
  const float* db        = (const float*)d_in[12];
  const float* Wa        = (const float*)d_in[13];  // [H, H]
  const float* Wc        = (const float*)d_in[14];  // [H, 2H]
  const float* bc        = (const float*)d_in[15];
  const float* Wout      = (const float*)d_in[16];  // [VT, H]
  const float* bout      = (const float*)d_in[17];
  float* out = (float*)d_out;                       // [T, B, VT]

  // ---- workspace carve-up (float units; every block is a multiple of 2048
  //      floats = 8KB, so all sub-buffers stay >=32B aligned) ----
  float* ws = (float*)d_ws;
  size_t off = 0;
  float* x_f   = ws + off; off += (size_t)SS*BB*4*HE;   // precomputed x@Wih_f + b
  float* x_b   = ws + off; off += (size_t)SS*BB*4*HE;
  float* enc   = ws + off; off += (size_t)BB*SS*HH;     // encoder outputs [B,S,H]
  float* keys  = ws + off; off += (size_t)BB*SS*HH;     // enc @ Wa        [B,S,H]
  float* dxe   = ws + off; off += (size_t)TT*BB*4*HH;   // emb_trg part of decoder gates
  float* h_f   = ws + off; off += (size_t)BB*HE;        // ---- zeroed region start ----
  float* c_f   = ws + off; off += (size_t)BB*HE;
  float* h_b   = ws + off; off += (size_t)BB*HE;
  float* c_b   = ws + off; off += (size_t)BB*HE;
  float* dh    = ws + off; off += (size_t)BB*HH;
  float* dc    = ws + off; off += (size_t)BB*HH;
  float* dctx  = ws + off; off += (size_t)BB*HH;        // ---- zeroed region end ----
  float* gates = ws + off; off += (size_t)BB*4*HH;
  float* score = ws + off; off += (size_t)BB*SS;
  float* cc    = ws + off; off += (size_t)BB*HH;
  int* idxf = (int*)(ws + off);
  int* idxb = idxf + SS*BB;
  int* idxd = idxb + SS*BB;
  off += 3*SS*BB;                                       // idx tables (int == float size)
  // optional bf16 staging (preferred roofline path)
  __bf16* wout16 = (__bf16*)(ws + off);
  const size_t wout16_f = ((size_t)VT*HH) / 2;          // halfs -> float units
  __bf16* cc16 = (__bf16*)(ws + off + wout16_f);
  const size_t need_full = (off + wout16_f + ((size_t)BB*HH)/2) * sizeof(float);
  const bool use_pre = (ws_size >= need_full);          // ws_size fixed -> deterministic

  const size_t zero_bytes = ((size_t)4*BB*HE + 3*(size_t)BB*HH) * sizeof(float);
  hipMemsetAsync(h_f, 0, zero_bytes, stream);
  build_idx<<<dim3((SS*BB + 255)/256), dim3(256), 0, stream>>>(src_batch, trg_batch, idxf, idxb, idxd);

  // one-time bf16 conversion of Wout (reused by all 64 decode steps, L2-resident)
  if (use_pre)
    f32_to_bf16<<<dim3(2048), dim3(256), 0, stream>>>(Wout, wout16, (long)VT*HH);

  // ---- hoisted input GEMMs (fused embedding gather) ----
  gemm_f32_wmma<<<dim3(4*HE/16, SS*BB/16), dim3(32), 0, stream>>>(
      emb_src, EE, idxf, eWih_f, EE, 0, 0, eb_f, nullptr, 0, x_f, 4*HE, EE, 0);
  gemm_f32_wmma<<<dim3(4*HE/16, SS*BB/16), dim3(32), 0, stream>>>(
      emb_src, EE, idxb, eWih_b, EE, 0, 0, eb_b, nullptr, 0, x_b, 4*HE, EE, 0);
  gemm_f32_wmma<<<dim3(4*HH/16, TT*BB/16), dim3(32), 0, stream>>>(
      emb_trg, EE, idxd, dWih, EE + HH, 0, 0, db, nullptr, 0, dxe, 4*HH, EE, 0);

  // ---- encoder recurrence (both directions) ----
  for (int s = 0; s < SS; ++s) {
    gemm_f32_wmma<<<dim3(4*HE/16, BB/16), dim3(32), 0, stream>>>(
        h_f, HE, nullptr, eWhh_f, HE, 0, 0, nullptr,
        x_f + (size_t)s*BB*4*HE, 4*HE, gates, 4*HE, HE, 0);
    lstm_cell<<<dim3((BB*HE + 255)/256), dim3(256), 0, stream>>>(
        gates, 4*HE, HE, h_f, c_f, enc + (size_t)s*HH, (long)SS*HH);
    gemm_f32_wmma<<<dim3(4*HE/16, BB/16), dim3(32), 0, stream>>>(
        h_b, HE, nullptr, eWhh_b, HE, 0, 0, nullptr,
        x_b + (size_t)s*BB*4*HE, 4*HE, gates, 4*HE, HE, 0);
    lstm_cell<<<dim3((BB*HE + 255)/256), dim3(256), 0, stream>>>(
        gates, 4*HE, HE, h_b, c_b, enc + (size_t)(SS - 1 - s)*HH + HE, (long)SS*HH);
  }

  // decoder initial state = concat(fwd, bwd)
  concat_hc<<<dim3((BB*HH + 255)/256), dim3(256), 0, stream>>>(h_f, h_b, dh);
  concat_hc<<<dim3((BB*HH + 255)/256), dim3(256), 0, stream>>>(c_f, c_b, dc);

  // attention keys: keys[b,s,:] = enc[b,s,:] @ Wa   (transposed W load)
  gemm_f32_wmma<<<dim3(HH/16, BB*SS/16), dim3(32), 0, stream>>>(
      enc, HH, nullptr, Wa, HH, 0, 1, nullptr, nullptr, 0, keys, HH, HH, 0);

  // ---- decoder recurrence ----
  for (int t = 0; t < TT; ++t) {
    gemm_f32_wmma<<<dim3(4*HH/16, BB/16), dim3(32), 0, stream>>>(
        dctx, HH, nullptr, dWih, EE + HH, EE, 0, nullptr,
        dxe + (size_t)t*BB*4*HH, 4*HH, gates, 4*HH, HH, 0);
    gemm_f32_wmma<<<dim3(4*HH/16, BB/16), dim3(32), 0, stream>>>(
        dh, HH, nullptr, dWhh, HH, 0, 0, nullptr,
        gates, 4*HH, gates, 4*HH, HH, 0);
    lstm_cell<<<dim3((BB*HH + 255)/256), dim3(256), 0, stream>>>(
        gates, 4*HH, HH, dh, dc, nullptr, 0);
    attn_score<<<dim3(BB*SS), dim3(32), 0, stream>>>(dh, keys, score);
    softmax_ctx<<<dim3(BB), dim3(256), 0, stream>>>(score, enc, dctx);
    // cc = tanh([ctx, h] @ Wc^T + bc)
    gemm_f32_wmma<<<dim3(HH/16, BB/16), dim3(32), 0, stream>>>(
        dctx, HH, nullptr, Wc, 2*HH, 0, 0, bc, nullptr, 0, cc, HH, HH, 0);
    gemm_f32_wmma<<<dim3(HH/16, BB/16), dim3(32), 0, stream>>>(
        dh, HH, nullptr, Wc, 2*HH, HH, 0, nullptr, cc, HH, cc, HH, HH, 1);
    // logits -> d_out slice for step t (bf16 WMMA, fp32 accumulate)
    if (use_pre) {
      f32_to_bf16<<<dim3(32), dim3(256), 0, stream>>>(cc, cc16, (long)BB*HH);
      outproj_bf16pre_wmma<<<dim3(VT/16, BB/16), dim3(32), 0, stream>>>(
          cc16, wout16, bout, out + (size_t)t*BB*VT, VT, HH);
    } else {
      outproj_bf16_wmma<<<dim3(VT/16, BB/16), dim3(32), 0, stream>>>(
          cc, Wout, bout, out + (size_t)t*BB*VT, VT, HH);
    }
  }

  // log_softmax over vocab for all T*B rows
  log_softmax_rows<<<dim3(TT*BB), dim3(256), 0, stream>>>(out, VT);
}
